// Linear4Bit_17102559773070
// MI455X (gfx1250) — matrix operand order repaired and checked by reading, not compiled, not run on hardware
//
#include <hip/hip_runtime.h>

typedef __attribute__((ext_vector_type(16))) _Float16 v16h;
typedef __attribute__((ext_vector_type(8)))  _Float16 v8h;
typedef __attribute__((ext_vector_type(8)))  float    v8f;
typedef __attribute__((ext_vector_type(4)))  float    v4f;
typedef __attribute__((ext_vector_type(4)))  int      v4i;

namespace {
constexpr int OUT_N  = 11008;
constexpr int IN_K   = 4096;
constexpr int TILE_M = 128;
constexpr int TILE_N = 128;
constexpr int TILE_K = 32;
constexpr int LDSW   = 40;                 // padded LDS row stride in halves (80B = 20 dwords, conflict-free)
constexpr int KTILES = IN_K / TILE_K;      // 128
constexpr int GPR    = IN_K / 16;          // 256 quant groups per output row
}

union FragAB { v16h v; v8h h[2]; };

__global__ __launch_bounds__(256)
void linear4bit_wmma_kernel(const float* __restrict__ x,
                            const int* __restrict__ wq4,
                            const _Float16* __restrict__ wnorm,
                            const float* __restrict__ bias,
                            float* __restrict__ out)
{
    __shared__ _Float16 Alds[2][TILE_M * LDSW];  // X tile as f16, [row][k]
    __shared__ _Float16 Blds[2][TILE_N * LDSW];  // W tile dequantized f16, [n][k]

    const int t    = threadIdx.x;
    const int lane = t & 31;
    const int wave = t >> 5;
    const int wm   = wave & 1;   // 2 waves along M -> 64 rows each
    const int wn   = wave >> 1;  // 4 waves along N -> 32 cols each

    const int bm = blockIdx.x * TILE_M;
    const int bn = blockIdx.y * TILE_N;

    // staging roles: thread t covers X row srow (half-row xcol) and W row srow (group gsel)
    const int srow = t >> 1;           // 0..127
    const int xcol = (t & 1) * 16;     // 0 or 16 within the 32-wide K tile
    const int gsel = t & 1;            // which group-of-16 within the K tile

    const float* xrow_ptr = x + (size_t)(bm + srow) * IN_K;
    const size_t gbase    = (size_t)(bn + srow) * GPR;

    v8f acc[4][2];
#pragma unroll
    for (int i = 0; i < 4; ++i)
#pragma unroll
        for (int j = 0; j < 2; ++j)
            acc[i][j] = {};

    // convert + dequantize staged registers into LDS buffer `buf`
    auto stage = [&](int buf, const v4f (&xv)[4], const v4i (&qv)[2], float nf) {
        // X fp32 -> f16
        v8h a0, a1;
#pragma unroll
        for (int j = 0; j < 4; ++j) {
            a0[j]     = (_Float16)xv[0][j];
            a0[4 + j] = (_Float16)xv[1][j];
            a1[j]     = (_Float16)xv[2][j];
            a1[4 + j] = (_Float16)xv[3][j];
        }
        _Float16* ap = &Alds[buf][srow * LDSW + xcol];
        *(v8h*)ap       = a0;
        *(v8h*)(ap + 8) = a1;

        // 4-bit dequant: w = norm*(2*code/15 - 1) = fma(code, norm*2/15, -norm)
        const float s = nf * (2.0f / 15.0f);
        v8h b0, b1;
#pragma unroll
        for (int p = 0; p < 4; ++p) {
            const int c0 = qv[0][p];
            const int c1 = qv[1][p];
            b0[2 * p]     = (_Float16)__builtin_fmaf((float)(c0 & 15),        s, -nf);
            b0[2 * p + 1] = (_Float16)__builtin_fmaf((float)((c0 >> 4) & 15), s, -nf);
            b1[2 * p]     = (_Float16)__builtin_fmaf((float)(c1 & 15),        s, -nf);
            b1[2 * p + 1] = (_Float16)__builtin_fmaf((float)((c1 >> 4) & 15), s, -nf);
        }
        _Float16* bp = &Blds[buf][srow * LDSW + gsel * 16];
        *(v8h*)bp       = b0;
        *(v8h*)(bp + 8) = b1;
    };

    // prologue: stage K-tile 0 into buffer 0
    v4f xv[4];
    v4i qv[2];
    float nf;
    {
        const float* p = xrow_ptr + xcol;
        xv[0] = *(const v4f*)(p);
        xv[1] = *(const v4f*)(p + 4);
        xv[2] = *(const v4f*)(p + 8);
        xv[3] = *(const v4f*)(p + 12);
        const size_t g = gbase + gsel;
        const v4i* qp  = (const v4i*)(wq4 + g * 8);
        qv[0] = qp[0];
        qv[1] = qp[1];
        nf    = (float)wnorm[g];
    }
    stage(0, xv, qv, nf);

    for (int kt = 0; kt < KTILES; ++kt) {
        __syncthreads();
        const int  buf  = kt & 1;
        const bool last = (kt == KTILES - 1);

        // prefetch next K-tile from global while WMMAs run on the current one
        if (!last) {
            const int    k0 = (kt + 1) * TILE_K;
            const float* p  = xrow_ptr + k0 + xcol;
            xv[0] = *(const v4f*)(p);
            xv[1] = *(const v4f*)(p + 4);
            xv[2] = *(const v4f*)(p + 8);
            xv[3] = *(const v4f*)(p + 12);
            const size_t g = gbase + (size_t)(k0 >> 4) + gsel;
            const v4i* qp  = (const v4i*)(wq4 + g * 8);
            qv[0] = qp[0];
            qv[1] = qp[1];
            nf    = (float)wnorm[g];
        }

        // A fragments: 16x32 f16, lanes 0-15 hold K 0-7 & 16-23, lanes 16-31 hold K 8-15 & 24-31
        FragAB a[4];
#pragma unroll
        for (int i = 0; i < 4; ++i) {
            const int row = wm * 64 + i * 16 + (lane & 15);
            const int kb  = (lane >> 4) * 8;
            const _Float16* pa = &Alds[buf][row * LDSW + kb];
            a[i].h[0] = *(const v8h*)pa;
            a[i].h[1] = *(const v8h*)(pa + 16);
        }
        // B fragments: 32x16 f16, lane n = lane%16, lanes 0-15 hold K 0-7 & 16-23, lanes 16-31 hold K 8-15 & 24-31
        FragAB b[2];
#pragma unroll
        for (int j = 0; j < 2; ++j) {
            const int n  = wn * 32 + j * 16 + (lane & 15);
            const int kb = (lane >> 4) * 8;
            const _Float16* pb = &Blds[buf][n * LDSW + kb];
            b[j].h[0] = *(const v8h*)pb;
            b[j].h[1] = *(const v8h*)(pb + 16);
        }

#pragma unroll
        for (int i = 0; i < 4; ++i)
#pragma unroll
            for (int j = 0; j < 2; ++j)
                acc[i][j] = __builtin_amdgcn_wmma_f32_16x16x32_f16(
                    false, a[i].v, false, b[j].v, (short)0, acc[i][j], false, false);

        if (!last) stage(buf ^ 1, xv, qv, nf);
    }

    // epilogue: C layout -> VGPR r holds M = r + 8*(lane>=16), N = lane%16
#pragma unroll
    for (int j = 0; j < 2; ++j) {
        const int   col = bn + wn * 32 + j * 16 + (lane & 15);
        const float bv  = bias[col];
#pragma unroll
        for (int i = 0; i < 4; ++i) {
            const int rowbase = bm + wm * 64 + i * 16 + (lane >> 4) * 8;
#pragma unroll
            for (int r = 0; r < 8; ++r)
                out[(size_t)(rowbase + r) * OUT_N + col] = acc[i][j][r] + bv;
        }
    }
}

extern "C" void kernel_launch(void* const* d_in, const int* in_sizes, int n_in,
                              void* d_out, int out_size, void* d_ws, size_t ws_size,
                              hipStream_t stream) {
    (void)n_in; (void)out_size; (void)d_ws; (void)ws_size;
    const float*     x    = (const float*)d_in[0];
    const int*       wq4  = (const int*)d_in[1];
    const _Float16*  wn   = (const _Float16*)d_in[2];
    const float*     bias = (const float*)d_in[3];
    float*           out  = (float*)d_out;

    const int M = in_sizes[0] / IN_K;  // 2*2048 = 4096
    dim3 grid(M / TILE_M, OUT_N / TILE_N);  // 32 x 86
    linear4bit_wmma_kernel<<<grid, 256, 0, stream>>>(x, wq4, wn, bias, out);
}
